// SpeakerGatedLocalTransformer_57028575756936
// MI455X (gfx1250) — matrix operand order repaired and checked by reading, not compiled, not run on hardware
//
#include <hip/hip_runtime.h>
#include <cstdint>
#include <cstddef>

// ---------------------------------------------------------------------------
// Types for CDNA5 WMMA (wave32, 16x16x32 bf16 -> f32)
// ---------------------------------------------------------------------------
typedef __bf16 bf16_t;
typedef __attribute__((ext_vector_type(8)))  __bf16 v8bf;
typedef __attribute__((ext_vector_type(16))) __bf16 v16bf;
typedef __attribute__((ext_vector_type(8)))  float  v8f;
typedef __attribute__((vector_size(16)))     int    vec4i;  // builtin's b128 type

#if __has_builtin(__builtin_amdgcn_global_load_async_to_lds_b128)
#define HAVE_ASYNC_LDS 1
#endif

__device__ __forceinline__ v16bf cat16(v8bf lo, v8bf hi) {
  return __builtin_shufflevector(lo, hi, 0,1,2,3,4,5,6,7,8,9,10,11,12,13,14,15);
}

// SRC0 (A) fragment: 16x32 bf16 tile at (row0, k0) from row-major LDS buffer.
__device__ __forceinline__ v16bf fragA(const bf16_t* s, int row0, int stride,
                                       int k0, int lane) {
  const bf16_t* p = s + (row0 + (lane & 15)) * stride + k0 + ((lane >> 4) << 3);
  return cat16(*(const v8bf*)p, *(const v8bf*)(p + 16));
}

// SRC1 (B) fragment: B_wmma[k][n] = s[row0+n][k] with s row-major [N][K].
__device__ __forceinline__ v16bf fragB(const bf16_t* s, int row0, int stride,
                                       int k0, int lane) {
  const bf16_t* p = s + (row0 + (lane & 15)) * stride + k0 + ((lane >> 4) << 3);
  return cat16(*(const v8bf*)p, *(const v8bf*)(p + 16));
}

__device__ __forceinline__ v8f wmma_bf16(v16bf a, v16bf b, v8f c) {
  return __builtin_amdgcn_wmma_f32_16x16x32_bf16(false, a, false, b,
                                                 (short)0, c, false, false);
}

#ifdef HAVE_ASYNC_LDS
// Per-lane 16B async copy global -> LDS (GLOBAL_LOAD_ASYNC_TO_LDS_B128,
// tracked by ASYNCcnt; no staging VGPRs).
__device__ __forceinline__ void async_cp16(const bf16_t* g, bf16_t* l) {
  __builtin_amdgcn_global_load_async_to_lds_b128(
      (__attribute__((address_space(1))) vec4i*)(g),
      (__attribute__((address_space(3))) vec4i*)(l), 0, 0);
}
__device__ __forceinline__ void wait_async() {
#if __has_builtin(__builtin_amdgcn_s_wait_asynccnt)
  __builtin_amdgcn_s_wait_asynccnt(0);
#else
  asm volatile("s_wait_asynccnt 0" ::: "memory");
#endif
}
#endif

// ---------------------------------------------------------------------------
// Generic GEMM: C[M,N] = A[M,K] @ B[N,K]^T + bias, A/B bf16 row-major.
// Block tile 128x256, 8 waves, 64x64 per wave (16 WMMA / wave / K-step).
// Double-buffered LDS, one barrier per K-step. Staging uses CDNA5 async
// global->LDS copies when available (fire-and-forget under the WMMAs).
// mode 0: f32 out; mode 1: exact-GELU, bf16 out; mode 2: bf16 out.
// Requires M%128==0, N%256==0, K%32==0.
// ---------------------------------------------------------------------------
__global__ __launch_bounds__(256) void gemm_bf16_kernel(
    const bf16_t* __restrict__ A, const bf16_t* __restrict__ B,
    const float* __restrict__ bias, float* __restrict__ Cf,
    bf16_t* __restrict__ Cb, int M, int N, int K, int mode)
{
  constexpr int STR = 40;  // 32 + 8 pad (keeps 16B alignment, spreads banks)
  __shared__ __align__(16) bf16_t As[2][128 * STR];  // 2 x 10240 B
  __shared__ __align__(16) bf16_t Bs[2][256 * STR];  // 2 x 20480 B
  const int tid  = threadIdx.x;
  const int lane = tid & 31;
  const int wave = tid >> 5;            // 8 waves
  const int wr = (wave & 1) * 64;       // wave row offset in 128x256 tile
  const int wc = (wave >> 1) * 64;      // wave col offset
  const long m0 = (long)blockIdx.y * 128;
  const long n0 = (long)blockIdx.x * 256;
  const int lrow = tid >> 1;            // 0..127
  const int lcol = (tid & 1) * 16;      // 0 or 16

  v8f acc[4][4] = {};

  const bf16_t* gA  = A + (m0 + lrow) * (long)K + lcol;
  const bf16_t* gB0 = B + (n0 + lrow) * (long)K + lcol;
  const bf16_t* gB1 = B + (n0 + 128 + lrow) * (long)K + lcol;

  const int sa0 = lrow * STR + lcol;          // LDS element offsets
  const int sb0 = lrow * STR + lcol;
  const int sb1 = (128 + lrow) * STR + lcol;

  // Prologue: stage K-step 0 into buffer 0.
#ifdef HAVE_ASYNC_LDS
  async_cp16(gA,      &As[0][sa0]);
  async_cp16(gA + 8,  &As[0][sa0 + 8]);
  async_cp16(gB0,     &Bs[0][sb0]);
  async_cp16(gB0 + 8, &Bs[0][sb0 + 8]);
  async_cp16(gB1,     &Bs[0][sb1]);
  async_cp16(gB1 + 8, &Bs[0][sb1 + 8]);
#else
  {
    uint4 ra0 = *(const uint4*)(gA);
    uint4 ra1 = *(const uint4*)(gA + 8);
    uint4 rb0 = *(const uint4*)(gB0);
    uint4 rb1 = *(const uint4*)(gB0 + 8);
    uint4 rb2 = *(const uint4*)(gB1);
    uint4 rb3 = *(const uint4*)(gB1 + 8);
    *(uint4*)&As[0][sa0]     = ra0;
    *(uint4*)&As[0][sa0 + 8] = ra1;
    *(uint4*)&Bs[0][sb0]     = rb0;
    *(uint4*)&Bs[0][sb0 + 8] = rb1;
    *(uint4*)&Bs[0][sb1]     = rb2;
    *(uint4*)&Bs[0][sb1 + 8] = rb3;
  }
#endif

  int buf = 0;
  for (int k0 = 0; k0 < K; k0 += 32) {
#ifdef HAVE_ASYNC_LDS
    wait_async();     // this wave's copies into `buf` are in LDS
#endif
    __syncthreads();  // buffer `buf` fully written; previous readers done
    const bool more = (k0 + 32) < K;
    const int nb = buf ^ 1;
#ifdef HAVE_ASYNC_LDS
    if (more) {       // fire-and-forget copies drain under the WMMAs
      async_cp16(gA + k0 + 32,  &As[nb][sa0]);
      async_cp16(gA + k0 + 40,  &As[nb][sa0 + 8]);
      async_cp16(gB0 + k0 + 32, &Bs[nb][sb0]);
      async_cp16(gB0 + k0 + 40, &Bs[nb][sb0 + 8]);
      async_cp16(gB1 + k0 + 32, &Bs[nb][sb1]);
      async_cp16(gB1 + k0 + 40, &Bs[nb][sb1 + 8]);
    }
#else
    uint4 ra0, ra1, rb0, rb1, rb2, rb3;
    if (more) {       // issue next-tile global loads early
      ra0 = *(const uint4*)(gA + k0 + 32);
      ra1 = *(const uint4*)(gA + k0 + 40);
      rb0 = *(const uint4*)(gB0 + k0 + 32);
      rb1 = *(const uint4*)(gB0 + k0 + 40);
      rb2 = *(const uint4*)(gB1 + k0 + 32);
      rb3 = *(const uint4*)(gB1 + k0 + 40);
    }
#endif
    const bf16_t* as = &As[buf][0];
    const bf16_t* bs = &Bs[buf][0];
    v16bf fa[4];
#pragma unroll
    for (int i = 0; i < 4; ++i) fa[i] = fragA(as, wr + i * 16, STR, 0, lane);
#pragma unroll
    for (int j = 0; j < 4; ++j) {
      v16bf fb = fragB(bs, wc + j * 16, STR, 0, lane);
#pragma unroll
      for (int i = 0; i < 4; ++i) acc[i][j] = wmma_bf16(fa[i], fb, acc[i][j]);
    }
#ifndef HAVE_ASYNC_LDS
    if (more) {       // fill the other buffer while WMMAs drain
      *(uint4*)&As[nb][sa0]     = ra0;
      *(uint4*)&As[nb][sa0 + 8] = ra1;
      *(uint4*)&Bs[nb][sb0]     = rb0;
      *(uint4*)&Bs[nb][sb0 + 8] = rb1;
      *(uint4*)&Bs[nb][sb1]     = rb2;
      *(uint4*)&Bs[nb][sb1 + 8] = rb3;
    }
#endif
    buf ^= 1;
  }

  const int rh = (lane >> 4) * 8;   // D layout: row = vgpr + 8*(lane/16)
  const int cn = lane & 15;         //           col = lane%16
#pragma unroll
  for (int i = 0; i < 4; ++i)
#pragma unroll
    for (int j = 0; j < 4; ++j) {
      long col = n0 + wc + j * 16 + cn;
      float bv = bias[col];
#pragma unroll
      for (int r = 0; r < 8; ++r) {
        long row = m0 + wr + i * 16 + rh + r;
        float v = acc[i][j][r] + bv;
        if (mode == 1) v = 0.5f * v * (1.0f + erff(v * 0.70710678118f));
        if (mode == 0) Cf[row * (long)N + col] = v;
        else           Cb[row * (long)N + col] = (bf16_t)v;
      }
    }
}

// ---------------------------------------------------------------------------
// Fused attention per (sequence, head). qkv: bf16 [65536, 2304].
// S = qk^T*scale + keymask, softmax in registers, O = P@V -> obf bf16.
// ---------------------------------------------------------------------------
__global__ __launch_bounds__(128) void attn_kernel(
    const bf16_t* __restrict__ qkv, const int* __restrict__ amask,
    bf16_t* __restrict__ obf)
{
  __shared__ __align__(16) char smem[50688];
  bf16_t* qs  = (bf16_t*)smem;            // 128 x 64
  bf16_t* ks  = (bf16_t*)(smem + 16384);  // 128 x 64
  bf16_t* vts = (bf16_t*)(smem + 32768);  // 64 x 136 (V transposed)
  float*  mb  = (float*)(smem + 50176);   // 128 key-bias
  bf16_t* ps  = (bf16_t*)smem;            // 128 x 128 P, overlays q+k

  const int n    = blockIdx.x;   // sequence 0..511
  const int hh   = blockIdx.y;   // head 0..11
  const int tid  = threadIdx.x;  // 128 threads = 4 waves
  const int lane = tid & 31;
  const int wave = tid >> 5;

  const bf16_t* qrow = qkv + (size_t)(n * 128 + tid) * 2304 + hh * 64;
#pragma unroll
  for (int c = 0; c < 64; c += 8) {
    *(uint4*)&qs[tid * 64 + c] = *(const uint4*)(qrow + c);
    *(uint4*)&ks[tid * 64 + c] = *(const uint4*)(qrow + 768 + c);
  }
#pragma unroll
  for (int d0 = 0; d0 < 64; d0 += 8) {
    v8bf vv = *(const v8bf*)(qrow + 1536 + d0);
#pragma unroll
    for (int u = 0; u < 8; ++u) vts[(d0 + u) * 136 + tid] = vv[u];
  }
  mb[tid] = amask[n * 128 + tid] ? 0.0f : -1.0e30f;
  __syncthreads();

  // S = q k^T  (each wave: 32 rows x 128 cols = 16 WMMA tiles)
  v8f sc[2][8] = {};
#pragma unroll
  for (int k0 = 0; k0 < 64; k0 += 32) {
    v16bf a0 = fragA(qs, wave * 32,      64, k0, lane);
    v16bf a1 = fragA(qs, wave * 32 + 16, 64, k0, lane);
#pragma unroll
    for (int j = 0; j < 8; ++j) {
      v16bf fb = fragB(ks, j * 16, 64, k0, lane);
      sc[0][j] = wmma_bf16(a0, fb, sc[0][j]);
      sc[1][j] = wmma_bf16(a1, fb, sc[1][j]);
    }
  }
  __syncthreads();  // all q/k LDS reads done; P overlay now safe

  // Register softmax: a D-layout row lives in one 16-lane half-wave.
  const int rh = (lane >> 4) * 8;
  const int cn = lane & 15;
  const float scale = 0.125f;  // 1/sqrt(64)
#pragma unroll
  for (int i = 0; i < 2; ++i) {
#pragma unroll
    for (int r = 0; r < 8; ++r) {
      float mx = -3.0e38f;
#pragma unroll
      for (int j = 0; j < 8; ++j) {
        float t = sc[i][j][r] * scale + mb[j * 16 + cn];
        sc[i][j][r] = t;
        mx = fmaxf(mx, t);
      }
#pragma unroll
      for (int o = 1; o < 16; o <<= 1) mx = fmaxf(mx, __shfl_xor(mx, o, 32));
      float sum = 0.0f;
#pragma unroll
      for (int j = 0; j < 8; ++j) {
        float e = __expf(sc[i][j][r] - mx);
        sc[i][j][r] = e;
        sum += e;
      }
#pragma unroll
      for (int o = 1; o < 16; o <<= 1) sum += __shfl_xor(sum, o, 32);
      float inv = 1.0f / sum;
      int row = wave * 32 + i * 16 + rh + r;
#pragma unroll
      for (int j = 0; j < 8; ++j)
        ps[row * 128 + j * 16 + cn] = (bf16_t)(sc[i][j][r] * inv);
    }
  }
  __syncthreads();

  // O = P @ V  (each wave: 32 rows x 64 dims = 8 WMMA tiles)
  v8f oc[2][4] = {};
#pragma unroll
  for (int k0 = 0; k0 < 128; k0 += 32) {
    v16bf a0 = fragA(ps, wave * 32,      128, k0, lane);
    v16bf a1 = fragA(ps, wave * 32 + 16, 128, k0, lane);
#pragma unroll
    for (int j = 0; j < 4; ++j) {
      v16bf fb = fragB(vts, j * 16, 136, k0, lane);
      oc[0][j] = wmma_bf16(a0, fb, oc[0][j]);
      oc[1][j] = wmma_bf16(a1, fb, oc[1][j]);
    }
  }
#pragma unroll
  for (int i = 0; i < 2; ++i)
#pragma unroll
    for (int j = 0; j < 4; ++j)
#pragma unroll
      for (int r = 0; r < 8; ++r) {
        int l = wave * 32 + i * 16 + rh + r;
        int d = j * 16 + cn;
        obf[(size_t)(n * 128 + l) * 768 + hh * 64 + d] = (bf16_t)oc[i][j][r];
      }
}

// ---------------------------------------------------------------------------
// y = LayerNorm(x + r) * g + b ; writes f32 master + bf16 GEMM copy.
// ---------------------------------------------------------------------------
__global__ __launch_bounds__(256) void add_ln_kernel(
    const float* __restrict__ x, const float* __restrict__ r,
    const float* __restrict__ g, const float* __restrict__ b,
    float* __restrict__ yf, bf16_t* __restrict__ ybf)
{
  __shared__ float red[256];
  const long row = blockIdx.x;
  const int tid = threadIdx.x;
  const float* xr = x + row * 768;
  const float* rr = r + row * 768;
  float v[3]; float s = 0.f;
#pragma unroll
  for (int t = 0; t < 3; ++t) { v[t] = xr[tid + t*256] + rr[tid + t*256]; s += v[t]; }
  red[tid] = s; __syncthreads();
  for (int o = 128; o > 0; o >>= 1) { if (tid < o) red[tid] += red[tid + o]; __syncthreads(); }
  float mean = red[0] * (1.0f / 768.0f);
  __syncthreads();
  float s2 = 0.f;
#pragma unroll
  for (int t = 0; t < 3; ++t) { float d = v[t] - mean; s2 += d * d; }
  red[tid] = s2; __syncthreads();
  for (int o = 128; o > 0; o >>= 1) { if (tid < o) red[tid] += red[tid + o]; __syncthreads(); }
  float rstd = rsqrtf(red[0] * (1.0f / 768.0f) + 1e-5f);
#pragma unroll
  for (int t = 0; t < 3; ++t) {
    int h = tid + t * 256;
    float o = (v[t] - mean) * rstd * g[h] + b[h];
    yf[row * 768 + h]  = o;
    ybf[row * 768 + h] = (bf16_t)o;
  }
}

// ---------------------------------------------------------------------------
// Embeddings: x = tok_emb[ids] + pos_emb[l] + spk_emb[spk[bt]]
// ---------------------------------------------------------------------------
__global__ __launch_bounds__(256) void embed_kernel(
    const int* __restrict__ ids, const int* __restrict__ spk,
    const float* __restrict__ te, const float* __restrict__ pe,
    const float* __restrict__ se, float* __restrict__ xf,
    bf16_t* __restrict__ xbf)
{
  const size_t i = (size_t)blockIdx.x * 256 + threadIdx.x;
  const int h = (int)(i % 768);
  const size_t row = i / 768;       // bt*128 + l
  const int l = (int)(row & 127);
  const size_t bt = row >> 7;
  float v = te[(size_t)ids[row] * 768 + h] + pe[l * 768 + h]
          + se[(size_t)spk[bt] * 768 + h];
  xf[i] = v; xbf[i] = (bf16_t)v;
}

__global__ __launch_bounds__(256) void conv_kernel(
    const float* __restrict__ s, bf16_t* __restrict__ d, long n)
{
  long i = (long)blockIdx.x * 256 + threadIdx.x;
  if (i < n) d[i] = (bf16_t)s[i];
}

// Masked mean pool over L per (sequence, channel-chunk)
__global__ __launch_bounds__(128) void pool_kernel(
    const float* __restrict__ tok, const int* __restrict__ amask,
    float* __restrict__ edu)
{
  const int n = blockIdx.x;
  const int c = blockIdx.y * 128 + threadIdx.x;
  const float* base = tok + (size_t)n * 128 * 768 + c;
  const int* mrow = amask + n * 128;
  float s = 0.f, cnt = 0.f;
  for (int l = 0; l < 128; ++l) {
    float m = mrow[l] ? 1.0f : 0.0f;
    s += m * base[(size_t)l * 768];
    cnt += m;
  }
  edu[n * 768 + c] = s / fmaxf(cnt, 1e-6f);
}

__global__ void last_kernel(const int* __restrict__ sids, int* __restrict__ last) {
  int tid = threadIdx.x;  // 512 threads, 1 block
  if (tid < 8) last[tid] = -1;
  __syncthreads();
  atomicMax(&last[sids[tid]], tid);
}

// GRU cell with hprev == 0:  mem[s] = (1-z)*tanh(inn + r*b_hh_n)
__global__ __launch_bounds__(128) void gru_kernel(
    const float* __restrict__ edu, const int* __restrict__ last,
    const float* __restrict__ wih, const float* __restrict__ bih,
    const float* __restrict__ bhh, float* __restrict__ mem)
{
  const int s = blockIdx.x;
  const int j = blockIdx.y * 128 + threadIdx.x;
  const int li = last[s];
  if (li < 0) { mem[s * 768 + j] = 0.f; return; }
  const float* e  = edu + (size_t)li * 768;
  const float* w0 = wih + (size_t)j * 768;
  const float* w1 = wih + (size_t)(768 + j) * 768;
  const float* w2 = wih + (size_t)(1536 + j) * 768;
  float ir = bih[j], iz = bih[768 + j], in_ = bih[1536 + j];
  for (int k = 0; k < 768; ++k) {
    float ev = e[k];
    ir += ev * w0[k]; iz += ev * w1[k]; in_ += ev * w2[k];
  }
  float rg = 1.f / (1.f + __expf(-(ir + bhh[j])));
  float zg = 1.f / (1.f + __expf(-(iz + bhh[768 + j])));
  float nn = tanhf(in_ + rg * bhh[1536 + j]);
  mem[s * 768 + j] = (1.f - zg) * nn;
}

// ---------------------------------------------------------------------------
// Host orchestration
// ---------------------------------------------------------------------------
extern "C" void kernel_launch(void* const* d_in, const int* in_sizes, int n_in,
                              void* d_out, int out_size, void* d_ws, size_t ws_size,
                              hipStream_t stream)
{
  (void)in_sizes; (void)n_in; (void)out_size; (void)ws_size;
  const int*   input_ids = (const int*)d_in[0];
  const int*   amask     = (const int*)d_in[1];
  const int*   spk       = (const int*)d_in[2];
  const float* te        = (const float*)d_in[3];
  const float* pe        = (const float*)d_in[4];
  const float* se        = (const float*)d_in[5];
  const float* qkv_w     = (const float*)d_in[6];
  const float* qkv_b     = (const float*)d_in[7];
  const float* out_w     = (const float*)d_in[8];
  const float* out_b     = (const float*)d_in[9];
  const float* w1        = (const float*)d_in[10];
  const float* b1        = (const float*)d_in[11];
  const float* w2        = (const float*)d_in[12];
  const float* b2        = (const float*)d_in[13];
  const float* ln1g      = (const float*)d_in[14];
  const float* ln1b      = (const float*)d_in[15];
  const float* ln2g      = (const float*)d_in[16];
  const float* ln2b      = (const float*)d_in[17];
  const float* gwih      = (const float*)d_in[18];
  const float* gbih      = (const float*)d_in[20];
  const float* gbhh      = (const float*)d_in[21];
  float* out = (float*)d_out;

  auto align = [](size_t x) { return (x + 255) & ~(size_t)255; };
  char* w = (char*)d_ws;
  size_t off = 0;
  float*  xf   = (float*)(w + off);  off = align(off + 65536UL * 768 * 4);
  bf16_t* xbf  = (bf16_t*)(w + off); off = align(off + 65536UL * 768 * 2);
  bf16_t* big  = (bf16_t*)(w + off); off = align(off + 65536UL * 3072 * 2); // qkv then ffn-h
  bf16_t* obf  = (bf16_t*)(w + off); off = align(off + 65536UL * 768 * 2);
  float*  rf   = (float*)(w + off);  off = align(off + 65536UL * 768 * 4);
  bf16_t* wqkv = (bf16_t*)(w + off); off = align(off + 2UL * 2304 * 768 * 2);
  bf16_t* wout = (bf16_t*)(w + off); off = align(off + 2UL * 768 * 768 * 2);
  bf16_t* w1b  = (bf16_t*)(w + off); off = align(off + 2UL * 3072 * 768 * 2);
  bf16_t* w2b  = (bf16_t*)(w + off); off = align(off + 2UL * 768 * 3072 * 2);
  float*  edu  = (float*)(w + off);  off = align(off + 512UL * 768 * 4);
  int*    last = (int*)(w + off);    off = align(off + 64);

  conv_kernel<<<(unsigned)((2UL*2304*768 + 255) / 256), 256, 0, stream>>>(qkv_w, wqkv, 2L*2304*768);
  conv_kernel<<<(unsigned)((2UL*768*768  + 255) / 256), 256, 0, stream>>>(out_w, wout, 2L*768*768);
  conv_kernel<<<(unsigned)((2UL*3072*768 + 255) / 256), 256, 0, stream>>>(w1, w1b, 2L*3072*768);
  conv_kernel<<<(unsigned)((2UL*768*3072 + 255) / 256), 256, 0, stream>>>(w2, w2b, 2L*768*3072);
  embed_kernel<<<196608, 256, 0, stream>>>(input_ids, spk, te, pe, se, xf, xbf);

  for (int layer = 0; layer < 2; ++layer) {
    gemm_bf16_kernel<<<dim3(9, 512), 256, 0, stream>>>(
        xbf, wqkv + (size_t)layer * 2304 * 768, qkv_b + layer * 2304,
        nullptr, big, 65536, 2304, 768, 2);
    attn_kernel<<<dim3(512, 12), 128, 0, stream>>>(big, amask, obf);
    gemm_bf16_kernel<<<dim3(3, 512), 256, 0, stream>>>(
        obf, wout + (size_t)layer * 768 * 768, out_b + layer * 768,
        rf, nullptr, 65536, 768, 768, 0);
    add_ln_kernel<<<65536, 256, 0, stream>>>(xf, rf, ln1g + layer * 768,
                                             ln1b + layer * 768, xf, xbf);
    gemm_bf16_kernel<<<dim3(12, 512), 256, 0, stream>>>(
        xbf, w1b + (size_t)layer * 3072 * 768, b1 + layer * 3072,
        nullptr, big, 65536, 3072, 768, 1);
    gemm_bf16_kernel<<<dim3(3, 512), 256, 0, stream>>>(
        big, w2b + (size_t)layer * 768 * 3072, b2 + layer * 768,
        rf, nullptr, 65536, 768, 3072, 0);
    float* dst = (layer == 1) ? out : xf;
    add_ln_kernel<<<65536, 256, 0, stream>>>(xf, rf, ln2g + layer * 768,
                                             ln2b + layer * 768, dst, xbf);
  }

  pool_kernel<<<dim3(512, 6), 128, 0, stream>>>(out, amask, edu);
  last_kernel<<<1, 512, 0, stream>>>(spk, last);
  gru_kernel<<<dim3(8, 6), 128, 0, stream>>>(edu, last, gwih, gbih, gbhh,
                                             out + 50331648UL);
}
